// DynamicAssembly_73091753443686
// MI455X (gfx1250) — matrix operand order repaired and checked by reading, not compiled, not run on hardware
//
#include <hip/hip_runtime.h>
#include <math.h>

#define BB 4
#define SS 1024
#define DD 1024
#define HH 16
#define MM 2
#define EE 8
#define KK 2
#define KDIM 64
#define FF 2048
#define HKD 1024  // H*KD

typedef __attribute__((ext_vector_type(16))) __bf16 v16bf;
typedef __attribute__((ext_vector_type(8)))  float  v8f;
typedef __attribute__((ext_vector_type(4)))  float  v4f;

// ---------- bf16 helpers ----------
static __device__ inline unsigned short bfbits(float f) {
  unsigned u = __builtin_bit_cast(unsigned, f);
  u = (u + 0x7FFFu + ((u >> 16) & 1u)) >> 16;
  return (unsigned short)u;
}
static __device__ inline __bf16 bfraw(unsigned short s) {
  return __builtin_bit_cast(__bf16, s);
}
static __device__ inline __bf16 f2bf(float f) { return bfraw(bfbits(f)); }

static __device__ inline v8f wmma_bf16(v16bf a, v16bf b, v8f c) {
  return __builtin_amdgcn_wmma_f32_16x16x32_bf16(false, a, false, b, (short)0, c,
                                                 false, false);
}

// Read one 16x16(x32) fragment (32 bytes / lane) from a fragment-major image.
static __device__ inline v16bf frag_at(const unsigned short* img, int blk, int lane) {
  return *reinterpret_cast<const v16bf*>(img + (((blk << 5) + lane) << 4));
}

// ---------- fragment-major staging (fp32 global -> bf16 LDS image) ----------
// A image: 8 row-blocks of 16 rows; per block, lane l = lh*16 + (m&15) holds
// 16 contiguous bf16 = k values {lh*8+0..7, lh*8+16..23} (ISA A-matrix layout).
// Thread t: row m = t>>1, k-half kh = t&1 -> 16 consecutive k (4x b128 global
// loads), written as two contiguous 16-byte LDS runs.
static __device__ inline void stage_a_128x32(unsigned short* As, const float* A,
                                             int lda, int row0, int kk, int t) {
  int m = t >> 1, kh = t & 1;
  const float* src = A + (size_t)(row0 + m) * lda + kk + kh * 16;
  v4f f0 = *(const v4f*)(src);
  v4f f1 = *(const v4f*)(src + 4);
  v4f f2 = *(const v4f*)(src + 8);
  v4f f3 = *(const v4f*)(src + 12);
  int blk = m >> 4, lo = m & 15;
  unsigned short* d0 = As + blk * 512 + lo * 16 + kh * 8;         // lh=0 half
  unsigned short* d1 = As + blk * 512 + (16 + lo) * 16 + kh * 8;  // lh=1 half
#pragma unroll
  for (int i = 0; i < 4; ++i) {
    d0[i]     = bfbits(f0[i]);
    d0[4 + i] = bfbits(f1[i]);
    d1[i]     = bfbits(f2[i]);
    d1[4 + i] = bfbits(f3[i]);
  }
}

// B image: 8 col-blocks of 16 cols; per block, lane l = lh*16 + (n&15) holds
// k values {lh*8+0..7, lh*8+16..23} for column n (ISA B-matrix layout).
// Thread t: k-row kr = t>>3, 16 consecutive n (4x b128 global loads), scattered
// 16-bit LDS stores (all inside one col-block).
static __device__ inline void stage_b_32x128(unsigned short* Bs, const float* B,
                                             int ldb, int kk, int col0, int t) {
  int kr = t >> 3, nstart = (t & 7) * 16;
  const float* src = B + (size_t)(kk + kr) * ldb + col0 + nstart;
  v4f f0 = *(const v4f*)(src);
  v4f f1 = *(const v4f*)(src + 4);
  v4f f2 = *(const v4f*)(src + 8);
  v4f f3 = *(const v4f*)(src + 12);
  int lanebase = (kr & 8) << 1;  // lh*16
  unsigned short* dst = Bs + (nstart >> 4) * 512 + ((kr & 7) | ((kr & 16) >> 1));
#pragma unroll
  for (int j = 0; j < 4; ++j) {
    dst[(lanebase + j) * 16]      = bfbits(f0[j]);
    dst[(lanebase + 4 + j) * 16]  = bfbits(f1[j]);
    dst[(lanebase + 8 + j) * 16]  = bfbits(f2[j]);
    dst[(lanebase + 12 + j) * 16] = bfbits(f3[j]);
  }
}

// 128x128 block tile, 256 threads = 8 waves, each wave 32x64 (2x4 fragments).
static __device__ inline void gemm_tile_128(const float* A, const float* Bm,
                                            int Kdim, int lda, int ldb,
                                            int row0, int col0,
                                            unsigned short* As, unsigned short* Bs,
                                            v8f acc[2][4]) {
  int t = threadIdx.x, lane = t & 31, w = t >> 5;
  int arb = ((w & 3) * 32) >> 4;   // A row-block base
  int bcb = ((w >> 2) * 64) >> 4;  // B col-block base
  for (int kk = 0; kk < Kdim; kk += 32) {
    __syncthreads();
    stage_a_128x32(As, A, lda, row0, kk, t);
    stage_b_32x128(Bs, Bm, ldb, kk, col0, t);
    __syncthreads();
    v16bf a0 = frag_at(As, arb, lane);
    v16bf a1 = frag_at(As, arb + 1, lane);
#pragma unroll
    for (int j = 0; j < 4; ++j) {
      v16bf bfj = frag_at(Bs, bcb + j, lane);
      acc[0][j] = wmma_bf16(a0, bfj, acc[0][j]);
      acc[1][j] = wmma_bf16(a1, bfj, acc[1][j]);
    }
  }
}

// ---------- LayerNorm + per-batch token-mean accumulation ----------
__global__ __launch_bounds__(256) void ln_kernel(
    const float* __restrict__ x, const float* __restrict__ g,
    const float* __restrict__ beta, float* __restrict__ nout,
    float* __restrict__ meansum) {
  int row = blockIdx.x;           // b*S + t
  int b = row >> 10;              // S = 1024
  int t = threadIdx.x;
  const float* xr = x + (size_t)row * DD;
  v4f xv = *(const v4f*)(xr + t * 4);   // each thread owns 4 elements in regs
  __shared__ float red[256];

  red[t] = xv[0] + xv[1] + xv[2] + xv[3];
  __syncthreads();
  for (int o = 128; o > 0; o >>= 1) { if (t < o) red[t] += red[t + o]; __syncthreads(); }
  float mean = red[0] * (1.0f / DD);
  __syncthreads();

  float vv = 0.f;
#pragma unroll
  for (int i = 0; i < 4; ++i) { float dv = xv[i] - mean; vv += dv * dv; }
  red[t] = vv; __syncthreads();
  for (int o = 128; o > 0; o >>= 1) { if (t < o) red[t] += red[t + o]; __syncthreads(); }
  float rstd = rsqrtf(red[0] * (1.0f / DD) + 1e-6f);

  v4f gv = *(const v4f*)(g + t * 4);
  v4f bv = *(const v4f*)(beta + t * 4);
  v4f ov;
#pragma unroll
  for (int i = 0; i < 4; ++i) ov[i] = (xv[i] - mean) * rstd * gv[i] + bv[i];
  *(v4f*)(nout + (size_t)row * DD + t * 4) = ov;
#pragma unroll
  for (int i = 0; i < 4; ++i)
    atomicAdd(&meansum[b * DD + t * 4 + i], ov[i] * (1.0f / SS));
}

// ---------- MHA router ----------
__global__ __launch_bounds__(256) void router_mha_kernel(
    const float* __restrict__ meansum, const float* __restrict__ rw,
    const float* __restrict__ rb, float* __restrict__ probs) {
  int b = blockIdx.x;
  int t = threadIdx.x;
  __shared__ float red[256];
  __shared__ float logits[MM];
  for (int m = 0; m < MM; ++m) {
    float s = 0.f;
    for (int d = t; d < DD; d += 256) s += meansum[b * DD + d] * rw[d * MM + m];
    red[t] = s; __syncthreads();
    for (int o = 128; o > 0; o >>= 1) { if (t < o) red[t] += red[t + o]; __syncthreads(); }
    if (t == 0) logits[m] = red[0] + rb[m];
    __syncthreads();
  }
  if (t == 0) {
    float mx = fmaxf(logits[0], logits[1]);
    float e0 = __expf(logits[0] - mx), e1 = __expf(logits[1] - mx);
    float sum = e0 + e1;
    probs[b * MM + 0] = e0 / sum;
    probs[b * MM + 1] = e1 / sum;
  }
}

// ---------- FFN router: softmax over E=8 + top-2 + renorm ----------
__global__ __launch_bounds__(256) void router_ffn_kernel(
    const float* __restrict__ meansum, const float* __restrict__ rw,
    const float* __restrict__ rb, int* __restrict__ topk_i,
    float* __restrict__ topk_p) {
  int b = blockIdx.x;
  int t = threadIdx.x;
  __shared__ float red[256];
  __shared__ float logits[EE];
  for (int e = 0; e < EE; ++e) {
    float s = 0.f;
    for (int d = t; d < DD; d += 256) s += meansum[b * DD + d] * rw[d * EE + e];
    red[t] = s; __syncthreads();
    for (int o = 128; o > 0; o >>= 1) { if (t < o) red[t] += red[t + o]; __syncthreads(); }
    if (t == 0) logits[e] = red[0] + rb[e];
    __syncthreads();
  }
  if (t == 0) {
    float mx = logits[0];
    for (int e = 1; e < EE; ++e) mx = fmaxf(mx, logits[e]);
    float pr[EE], sum = 0.f;
    for (int e = 0; e < EE; ++e) { pr[e] = __expf(logits[e] - mx); sum += pr[e]; }
    for (int e = 0; e < EE; ++e) pr[e] /= sum;
    int i0 = 0;
    for (int e = 1; e < EE; ++e) if (pr[e] > pr[i0]) i0 = e;
    int i1 = (i0 == 0) ? 1 : 0;
    for (int e = 0; e < EE; ++e) if (e != i0 && pr[e] > pr[i1]) i1 = e;
    float p0 = pr[i0], p1 = pr[i1], s2 = p0 + p1;
    topk_i[b * KK + 0] = i0; topk_i[b * KK + 1] = i1;
    topk_p[b * KK + 0] = p0 / s2; topk_p[b * KK + 1] = p1 / s2;
  }
}

// ---------- GEMM: C = A @ B + bias ----------
__global__ __launch_bounds__(256) void gemm_bias_kernel(
    const float* __restrict__ A, const float* __restrict__ Bm,
    const float* __restrict__ bias, float* __restrict__ C,
    int Kdim, int lda, int ldb, int ldc) {
  __shared__ __align__(32) unsigned short As[4096];
  __shared__ __align__(32) unsigned short Bs[4096];
  int t = threadIdx.x, lane = t & 31, w = t >> 5;
  int row0 = blockIdx.y * 128, col0 = blockIdx.x * 128;
  int wr = (w & 3) * 32, wc = (w >> 2) * 64;
  v8f acc[2][4] = {};
  gemm_tile_128(A, Bm, Kdim, lda, ldb, row0, col0, As, Bs, acc);
  int rsel = (lane >> 4) * 8, nsel = lane & 15;
#pragma unroll
  for (int ti = 0; ti < 2; ++ti)
#pragma unroll
    for (int tj = 0; tj < 4; ++tj) {
      int col = col0 + wc + tj * 16 + nsel;
      float bv = bias[col];
#pragma unroll
      for (int r = 0; r < 8; ++r) {
        int row = row0 + wr + ti * 16 + rsel + r;
        C[(size_t)row * ldc + col] = acc[ti][tj][r] + bv;
      }
    }
}

// ---------- flash attention over one (m,b,h), 64 query rows / block ----------
__global__ __launch_bounds__(128) void attn_kernel(
    const float* __restrict__ q, const float* __restrict__ k,
    const float* __restrict__ v, float* __restrict__ ctx) {
  // B-fragment images for K^T: [kdstep*2 + keyblock][lane][16]
  __shared__ __align__(32) unsigned short Kimg[4][512];
  // B-fragment images for V: [kd colblock][lane][16]
  __shared__ __align__(32) unsigned short Vimg[4][512];
  // per-wave A-fragment image for P
  __shared__ __align__(32) unsigned short Pimg[4][512];
  int t = threadIdx.x, lane = t & 31, w = t >> 5;
  int mb = blockIdx.z;
  int h = blockIdx.y;
  int q0 = blockIdx.x * 64 + w * 16;
  size_t base = (size_t)mb * SS * HKD + (size_t)h * KDIM;
  const float* Q = q + base;
  const float* Kg = k + base;
  const float* Vg = v + base;

  // Q A-fragments (kd 0..31 and 32..63), softmax scale 1/8 folded in
  v16bf aq0, aq1;
  {
    int mr = q0 + (lane & 15);
    int kb = (lane >> 4) * 8;
    const float* src = Q + (size_t)mr * HKD + kb;
    v4f a = *(const v4f*)(src);       v4f b2 = *(const v4f*)(src + 4);
    v4f c = *(const v4f*)(src + 16);  v4f d = *(const v4f*)(src + 20);
    v4f e = *(const v4f*)(src + 32);  v4f f = *(const v4f*)(src + 36);
    v4f g2 = *(const v4f*)(src + 48); v4f h2 = *(const v4f*)(src + 52);
#pragma unroll
    for (int i = 0; i < 4; ++i) {
      aq0[i]      = f2bf(a[i] * 0.125f);
      aq0[4 + i]  = f2bf(b2[i] * 0.125f);
      aq0[8 + i]  = f2bf(c[i] * 0.125f);
      aq0[12 + i] = f2bf(d[i] * 0.125f);
      aq1[i]      = f2bf(e[i] * 0.125f);
      aq1[4 + i]  = f2bf(f[i] * 0.125f);
      aq1[8 + i]  = f2bf(g2[i] * 0.125f);
      aq1[12 + i] = f2bf(h2[i] * 0.125f);
    }
  }

  float mrow[8], lrow[8];
#pragma unroll
  for (int r = 0; r < 8; ++r) { mrow[r] = -1e30f; lrow[r] = 0.f; }
  v8f o[4] = {};

  for (int kt = 0; kt < SS; kt += 32) {
    __syncthreads();
    // stage K^T B-fragment images (contiguous reads + contiguous stores)
    {
      int img = t >> 5, l = t & 31;
      int s = img >> 1, kb2 = img & 1;
      int key = kb2 * 16 + (l & 15);
      int kd0 = s * 32 + ((l >> 4) & 1) * 8;
      const float* src = Kg + (size_t)(kt + key) * HKD + kd0;
      v4f f0 = *(const v4f*)(src);
      v4f f1 = *(const v4f*)(src + 4);
      v4f f2 = *(const v4f*)(src + 16);
      v4f f3 = *(const v4f*)(src + 20);
      unsigned short* dst = &Kimg[img][l * 16];
#pragma unroll
      for (int i = 0; i < 4; ++i) {
        dst[i]      = bfbits(f0[i]);
        dst[4 + i]  = bfbits(f1[i]);
        dst[8 + i]  = bfbits(f2[i]);
        dst[12 + i] = bfbits(f3[i]);
      }
    }
    // stage V B-fragment images (contiguous reads along kd, scattered stores)
    {
      int key = t >> 2, kds = (t & 3) * 16;
      const float* src = Vg + (size_t)(kt + key) * HKD + kds;
      v4f f0 = *(const v4f*)(src);
      v4f f1 = *(const v4f*)(src + 4);
      v4f f2 = *(const v4f*)(src + 8);
      v4f f3 = *(const v4f*)(src + 12);
      int lanebase = (key & 8) << 1;  // lh*16
      unsigned short* dst = &Vimg[kds >> 4][(key & 7) | ((key & 16) >> 1)];
#pragma unroll
      for (int j = 0; j < 4; ++j) {
        dst[(lanebase + j) * 16]      = bfbits(f0[j]);
        dst[(lanebase + 4 + j) * 16]  = bfbits(f1[j]);
        dst[(lanebase + 8 + j) * 16]  = bfbits(f2[j]);
        dst[(lanebase + 12 + j) * 16] = bfbits(f3[j]);
      }
    }
    __syncthreads();

    // scores
    v8f s0 = {}, s1 = {};
    s0 = wmma_bf16(aq0, *reinterpret_cast<const v16bf*>(&Kimg[0][lane * 16]), s0);
    s0 = wmma_bf16(aq1, *reinterpret_cast<const v16bf*>(&Kimg[2][lane * 16]), s0);
    s1 = wmma_bf16(aq0, *reinterpret_cast<const v16bf*>(&Kimg[1][lane * 16]), s1);
    s1 = wmma_bf16(aq1, *reinterpret_cast<const v16bf*>(&Kimg[3][lane * 16]), s1);

    // online softmax (row stats live across 16-lane halves)
    float p0[8], p1[8];
#pragma unroll
    for (int r = 0; r < 8; ++r) {
      float mx = fmaxf(s0[r], s1[r]);
#pragma unroll
      for (int off = 1; off < 16; off <<= 1) mx = fmaxf(mx, __shfl_xor(mx, off, 32));
      float nm = fmaxf(mrow[r], mx);
      float alpha = __expf(mrow[r] - nm);
      mrow[r] = nm;
      float e0 = __expf(s0[r] - nm), e1 = __expf(s1[r] - nm);
      float rs = e0 + e1;
#pragma unroll
      for (int off = 1; off < 16; off <<= 1) rs += __shfl_xor(rs, off, 32);
      lrow[r] = lrow[r] * alpha + rs;
      p0[r] = e0; p1[r] = e1;
#pragma unroll
      for (int j = 0; j < 4; ++j) o[j][r] = o[j][r] * alpha;
    }

    // scatter P (C-layout) directly into A-fragment image
    {
      int n = lane & 15, rbase = (lane >> 4) * 8;
      int lh = (n >> 3) & 1, io = n & 7;
#pragma unroll
      for (int r = 0; r < 8; ++r) {
        int m = rbase + r;
        Pimg[w][(lh * 16 + m) * 16 + io]     = bfbits(p0[r]);
        Pimg[w][(lh * 16 + m) * 16 + 8 + io] = bfbits(p1[r]);
      }
    }
    __syncthreads();
    v16bf ap = *reinterpret_cast<const v16bf*>(&Pimg[w][lane * 16]);
#pragma unroll
    for (int j = 0; j < 4; ++j)
      o[j] = wmma_bf16(ap, *reinterpret_cast<const v16bf*>(&Vimg[j][lane * 16]), o[j]);
  }

  int rbase = (lane >> 4) * 8, n = lane & 15;
#pragma unroll
  for (int j = 0; j < 4; ++j)
#pragma unroll
    for (int r = 0; r < 8; ++r) {
      int row = q0 + rbase + r;
      ctx[base + (size_t)row * HKD + j * 16 + n] = o[j][r] / lrow[r];
    }
}

// ---------- out-proj + weighted combine: x1 = x + sum_m p*(ctx_m@Wo_m+bo_m)
__global__ __launch_bounds__(256) void attn_out_kernel(
    const float* __restrict__ ctx, const float* __restrict__ wo,
    const float* __restrict__ bo, const float* __restrict__ probs,
    const float* __restrict__ x, float* __restrict__ x1) {
  __shared__ __align__(32) unsigned short As[4096];
  __shared__ __align__(32) unsigned short Bs[4096];
  int t = threadIdx.x, lane = t & 31, w = t >> 5;
  int row0 = blockIdx.y * 128, col0 = blockIdx.x * 128;
  int b = row0 >> 10;  // 128-row tile stays within one batch (S=1024)
  int wr = (w & 3) * 32, wc = (w >> 2) * 64;
  int rsel = (lane >> 4) * 8, nsel = lane & 15;
  float tot[2][4][8] = {};

  for (int m = 0; m < MM; ++m) {
    float p = probs[b * MM + m];
    const float* A = ctx + (size_t)m * BB * SS * HKD;
    const float* Bm = wo + (size_t)m * HKD * DD;
    const float* bias = bo + m * DD;
    v8f acc[2][4] = {};
    gemm_tile_128(A, Bm, HKD, HKD, DD, row0, col0, As, Bs, acc);
#pragma unroll
    for (int ti = 0; ti < 2; ++ti)
#pragma unroll
      for (int tj = 0; tj < 4; ++tj) {
        float bv = bias[col0 + wc + tj * 16 + nsel];
#pragma unroll
        for (int r = 0; r < 8; ++r)
          tot[ti][tj][r] += p * (acc[ti][tj][r] + bv);
      }
  }
#pragma unroll
  for (int ti = 0; ti < 2; ++ti)
#pragma unroll
    for (int tj = 0; tj < 4; ++tj)
#pragma unroll
      for (int r = 0; r < 8; ++r) {
        int row = row0 + wr + ti * 16 + rsel + r;
        int col = col0 + wc + tj * 16 + nsel;
        x1[(size_t)row * DD + col] = x[(size_t)row * DD + col] + tot[ti][tj][r];
      }
}

// ---------- FFN1: h[b,slot] = gelu(n2[b] @ fw1[e]) ----------
__global__ __launch_bounds__(256) void ffn1_kernel(
    const float* __restrict__ n2, const float* __restrict__ fw1,
    const int* __restrict__ topk_i, float* __restrict__ h) {
  __shared__ __align__(32) unsigned short As[4096];
  __shared__ __align__(32) unsigned short Bs[4096];
  int t = threadIdx.x, lane = t & 31, w = t >> 5;
  int z = blockIdx.z;                 // b*K + slot
  int b = z / KK;
  int e = topk_i[z];
  const float* A = n2 + (size_t)b * SS * DD;
  const float* Bm = fw1 + (size_t)e * DD * FF;
  float* C = h + (size_t)z * SS * FF;
  int row0 = blockIdx.y * 128, col0 = blockIdx.x * 128;
  int wr = (w & 3) * 32, wc = (w >> 2) * 64;
  v8f acc[2][4] = {};
  gemm_tile_128(A, Bm, DD, DD, FF, row0, col0, As, Bs, acc);
  int rsel = (lane >> 4) * 8, nsel = lane & 15;
#pragma unroll
  for (int ti = 0; ti < 2; ++ti)
#pragma unroll
    for (int tj = 0; tj < 4; ++tj) {
      int col = col0 + wc + tj * 16 + nsel;
#pragma unroll
      for (int r = 0; r < 8; ++r) {
        int row = row0 + wr + ti * 16 + rsel + r;
        float vv = acc[ti][tj][r];
        vv = 0.5f * vv * (1.0f + erff(vv * 0.70710678118f));  // exact GELU
        C[(size_t)row * FF + col] = vv;
      }
    }
}

// ---------- FFN2 + combine: out = x1 + sum_slot p*(h[b,slot] @ fw2[e]) ----------
__global__ __launch_bounds__(256) void ffn2_kernel(
    const float* __restrict__ h, const float* __restrict__ fw2,
    const int* __restrict__ topk_i, const float* __restrict__ topk_p,
    const float* __restrict__ x1, float* __restrict__ out) {
  __shared__ __align__(32) unsigned short As[4096];
  __shared__ __align__(32) unsigned short Bs[4096];
  int t = threadIdx.x, lane = t & 31, w = t >> 5;
  int b = blockIdx.z;
  int row0 = blockIdx.y * 128, col0 = blockIdx.x * 128;
  int wr = (w & 3) * 32, wc = (w >> 2) * 64;
  int rsel = (lane >> 4) * 8, nsel = lane & 15;
  float tot[2][4][8] = {};

  for (int slot = 0; slot < KK; ++slot) {
    int e = topk_i[b * KK + slot];
    float p = topk_p[b * KK + slot];
    const float* A = h + (size_t)(b * KK + slot) * SS * FF;
    const float* Bm = fw2 + (size_t)e * FF * DD;
    v8f acc[2][4] = {};
    gemm_tile_128(A, Bm, FF, FF, DD, row0, col0, As, Bs, acc);
#pragma unroll
    for (int ti = 0; ti < 2; ++ti)
#pragma unroll
      for (int tj = 0; tj < 4; ++tj)
#pragma unroll
        for (int r = 0; r < 8; ++r) tot[ti][tj][r] += p * acc[ti][tj][r];
  }
#pragma unroll
  for (int ti = 0; ti < 2; ++ti)
#pragma unroll
    for (int tj = 0; tj < 4; ++tj)
#pragma unroll
      for (int r = 0; r < 8; ++r) {
        int srow = row0 + wr + ti * 16 + rsel + r;     // within S
        int col = col0 + wc + tj * 16 + nsel;
        size_t gidx = ((size_t)b * SS + srow) * DD + col;
        out[gidx] = x1[gidx] + tot[ti][tj][r];
      }
}

// ---------- host ----------
extern "C" void kernel_launch(void* const* d_in, const int* in_sizes, int n_in,
                              void* d_out, int out_size, void* d_ws, size_t ws_size,
                              hipStream_t stream) {
  const float* x     = (const float*)d_in[0];
  const float* ln1_g = (const float*)d_in[1];
  const float* ln1_b = (const float*)d_in[2];
  const float* ln2_g = (const float*)d_in[3];
  const float* ln2_b = (const float*)d_in[4];
  const float* wq    = (const float*)d_in[5];
  const float* bq    = (const float*)d_in[6];
  const float* wk    = (const float*)d_in[7];
  const float* bk    = (const float*)d_in[8];
  const float* wv    = (const float*)d_in[9];
  const float* bv    = (const float*)d_in[10];
  const float* wo    = (const float*)d_in[11];
  const float* bo    = (const float*)d_in[12];
  const float* rw    = (const float*)d_in[13];
  const float* rb    = (const float*)d_in[14];
  const float* frw   = (const float*)d_in[15];
  const float* frb   = (const float*)d_in[16];
  const float* fw1   = (const float*)d_in[17];
  const float* fw2   = (const float*)d_in[18];
  float* out = (float*)d_out;

  const size_t NTOK = (size_t)BB * SS;          // 4096
  char* ws = (char*)d_ws;
  size_t off = 0;
  auto alloc = [&](size_t nfloats) {
    float* p = (float*)(ws + off);
    off += nfloats * sizeof(float);
    return p;
  };
  float* nbuf  = alloc(NTOK * DD);              // n1, later reused as n2
  float* qbuf  = alloc((size_t)MM * NTOK * HKD);
  float* kbuf  = alloc((size_t)MM * NTOK * HKD);
  float* vbuf  = alloc((size_t)MM * NTOK * HKD);
  float* cbuf  = alloc((size_t)MM * NTOK * HKD);
  float* x1buf = alloc(NTOK * DD);
  float* meanb = alloc((size_t)BB * DD);
  float* probs = alloc(BB * MM);
  float* tkp   = alloc(BB * KK);
  int*   tki   = (int*)alloc(BB * KK);
  float* hbuf  = qbuf;  // h [B*K, S, F] (64MB) overlays dead q+k regions (64MB)

  // ---- phase 1: LN1 + routing stats ----
  hipMemsetAsync(meanb, 0, (size_t)BB * DD * sizeof(float), stream);
  ln_kernel<<<dim3(NTOK), dim3(256), 0, stream>>>(x, ln1_g, ln1_b, nbuf, meanb);
  router_mha_kernel<<<dim3(BB), dim3(256), 0, stream>>>(meanb, rw, rb, probs);

  // ---- QKV projections (bf16 WMMA GEMMs) ----
  dim3 gqkv(HKD / 128, NTOK / 128);
  for (int m = 0; m < MM; ++m) {
    gemm_bias_kernel<<<gqkv, 256, 0, stream>>>(
        nbuf, wq + (size_t)m * DD * HKD, bq + m * HKD,
        qbuf + (size_t)m * NTOK * HKD, DD, DD, HKD, HKD);
    gemm_bias_kernel<<<gqkv, 256, 0, stream>>>(
        nbuf, wk + (size_t)m * DD * HKD, bk + m * HKD,
        kbuf + (size_t)m * NTOK * HKD, DD, DD, HKD, HKD);
    gemm_bias_kernel<<<gqkv, 256, 0, stream>>>(
        nbuf, wv + (size_t)m * DD * HKD, bv + m * HKD,
        vbuf + (size_t)m * NTOK * HKD, DD, DD, HKD, HKD);
  }

  // ---- flash attention ----
  attn_kernel<<<dim3(SS / 64, HH, MM * BB), 128, 0, stream>>>(qbuf, kbuf, vbuf, cbuf);

  // ---- output projection + weighted combine + residual ----
  attn_out_kernel<<<dim3(DD / 128, NTOK / 128), 256, 0, stream>>>(
      cbuf, wo, bo, probs, x, x1buf);

  // ---- phase 2: LN2 + FFN routing ----
  hipMemsetAsync(meanb, 0, (size_t)BB * DD * sizeof(float), stream);
  ln_kernel<<<dim3(NTOK), dim3(256), 0, stream>>>(x1buf, ln2_g, ln2_b, nbuf, meanb);
  router_ffn_kernel<<<dim3(BB), dim3(256), 0, stream>>>(meanb, frw, frb, tki, tkp);

  // ---- FFN expert GEMMs ----
  ffn1_kernel<<<dim3(FF / 128, SS / 128, BB * KK), 256, 0, stream>>>(nbuf, fw1, tki, hbuf);
  ffn2_kernel<<<dim3(DD / 128, SS / 128, BB), 256, 0, stream>>>(hbuf, fw2, tki, tkp,
                                                                x1buf, out);
}
